// PointNet_SD_38208029065493
// MI455X (gfx1250) — compile-verified, matrix operand order checked by reading, not run on hardware
//
#include <hip/hip_runtime.h>
#include <hip/hip_bf16.h>
#include <math.h>

// ---------------------------------------------------------------------------
// PointNet++ (3-branch) for MI455X / gfx1250.
// Matrix work -> v_wmma_f32_16x16x32_f16 (f16 in, f32 accum), register-blocked
// 16x64 output tile per wave (A-fragment reuse x4 => 4x less activation
// re-read; activations dominate traffic at 23.3 TB/s HBM + 192 MB L2).
// Irregular work (FPS / kNN / gather) -> VALU + LDS.
// ---------------------------------------------------------------------------

typedef _Float16 v16h __attribute__((ext_vector_type(16)));
typedef _Float16 v8h  __attribute__((ext_vector_type(8)));
typedef float    v8f  __attribute__((ext_vector_type(8)));

// ------------------------------- transpose ---------------------------------
// (B,3,N) -> (B,N,3)
__global__ void pn_transpose_kernel(const float* __restrict__ pc,
                                    float* __restrict__ xyz, int B, int N) {
  long long gid = (long long)blockIdx.x * blockDim.x + threadIdx.x;
  long long total = (long long)B * N * 3;
  if (gid >= total) return;
  int c = (int)(gid % 3);
  long long bn = gid / 3;
  int n = (int)(bn % N);
  int b = (int)(bn / N);
  xyz[gid] = pc[((long long)b * 3 + c) * N + n];
}

// --------------------------------- FPS -------------------------------------
// One block per batch. Distance array lives in LDS (N <= 2048 -> 8 KB of the
// 320 KB WGP pool). Sequential S-step scan with block-wide argmax reduction.
__global__ void pn_fps_kernel(const float* __restrict__ xyz, int N, int S,
                              int* __restrict__ fidx, float* __restrict__ nxyz) {
  __shared__ float sdist[2048];
  __shared__ float smax[256];
  __shared__ int   simax[256];
  __shared__ int   slast;

  int b = blockIdx.x;
  int tid = threadIdx.x;
  const float* xb = xyz + (long long)b * N * 3;

  for (int n = tid; n < N; n += blockDim.x) sdist[n] = 1e10f;
  if (tid == 0) {
    slast = 0;
    fidx[(long long)b * S] = 0;
    nxyz[((long long)b * S) * 3 + 0] = xb[0];
    nxyz[((long long)b * S) * 3 + 1] = xb[1];
    nxyz[((long long)b * S) * 3 + 2] = xb[2];
  }
  __syncthreads();

  for (int it = 1; it < S; ++it) {
    int last = slast;
    float lx = xb[last * 3 + 0], ly = xb[last * 3 + 1], lz = xb[last * 3 + 2];
    float bmax = -1.0f; int bidx = 0;
    for (int n = tid; n < N; n += blockDim.x) {
      float dx = xb[n * 3 + 0] - lx;
      float dy = xb[n * 3 + 1] - ly;
      float dz = xb[n * 3 + 2] - lz;
      float d = dx * dx + dy * dy + dz * dz;
      float cur = fminf(sdist[n], d);
      sdist[n] = cur;
      if (cur > bmax) { bmax = cur; bidx = n; }
    }
    smax[tid] = bmax; simax[tid] = bidx;
    __syncthreads();
    for (int s = blockDim.x >> 1; s > 0; s >>= 1) {
      if (tid < s && smax[tid + s] > smax[tid]) {
        smax[tid] = smax[tid + s]; simax[tid] = simax[tid + s];
      }
      __syncthreads();
    }
    if (tid == 0) {
      int nx = simax[0];
      slast = nx;
      fidx[(long long)b * S + it] = nx;
      nxyz[((long long)b * S + it) * 3 + 0] = xb[nx * 3 + 0];
      nxyz[((long long)b * S + it) * 3 + 1] = xb[nx * 3 + 1];
      nxyz[((long long)b * S + it) * 3 + 2] = xb[nx * 3 + 2];
    }
    __syncthreads();
  }
}

// --------------------------------- kNN -------------------------------------
// One thread per (b, s) center; insertion top-K (K <= 48).
__global__ void pn_knn_kernel(const float* __restrict__ xyz,
                              const float* __restrict__ ctr,
                              int B, int N, int S, int K,
                              int* __restrict__ idx) {
  int gid = blockIdx.x * blockDim.x + threadIdx.x;
  if (gid >= B * S) return;
  int b = gid / S;
  const float* xb = xyz + (long long)b * N * 3;
  float cx = ctr[(long long)gid * 3 + 0];
  float cy = ctr[(long long)gid * 3 + 1];
  float cz = ctr[(long long)gid * 3 + 2];

  float bd[48]; int bi[48];
  for (int j = 0; j < K; ++j) { bd[j] = 3e38f; bi[j] = 0; }
  for (int n = 0; n < N; ++n) {
    float dx = xb[n * 3 + 0] - cx;
    float dy = xb[n * 3 + 1] - cy;
    float dz = xb[n * 3 + 2] - cz;
    float d = dx * dx + dy * dy + dz * dz;
    if (d < bd[K - 1]) {
      int j = K - 1;
      while (j > 0 && bd[j - 1] > d) { bd[j] = bd[j - 1]; bi[j] = bi[j - 1]; --j; }
      bd[j] = d; bi[j] = n;
    }
  }
  int* o = idx + (long long)gid * K;
  for (int j = 0; j < K; ++j) o[j] = bi[j];
}

// ---------------------------- gather + concat ------------------------------
// Builds X (B*S*K rows x Cpad cols, f16): [xyz(-center) | feat | zero-pad].
// idx==NULL -> identity neighbor (group_all); ctr==NULL -> no re-centering.
__global__ void pn_gather_kernel(const float* __restrict__ xyz,
                                 const float* __restrict__ feat,
                                 const int* __restrict__ idx,
                                 const float* __restrict__ ctr,
                                 _Float16* __restrict__ X,
                                 int B, int S, int K, int N, int C, int Cpad) {
  long long gid = (long long)blockIdx.x * blockDim.x + threadIdx.x;
  long long total = (long long)B * S * K * Cpad;
  if (gid >= total) return;
  int c = (int)(gid % Cpad);
  long long r = gid / Cpad;
  int j = (int)(r % K);
  long long bs = r / K;
  int s = (int)(bs % S);
  int b = (int)(bs / S);
  int n = idx ? idx[((long long)b * S + s) * K + j] : j;
  float v = 0.0f;
  if (c < 3) {
    v = xyz[((long long)b * N + n) * 3 + c];
    if (ctr) v -= ctr[((long long)b * S + s) * 3 + c];
  } else if (c < 3 + C) {
    v = feat[((long long)b * N + n) * C + (c - 3)];
  }
  X[gid] = (_Float16)v;
}

// ------------------------- f32 -> f16 pad/convert --------------------------
__global__ void pn_pad_f16_kernel(const float* __restrict__ W,
                                  _Float16* __restrict__ Wh,
                                  int O, int C, int Cpad) {
  int gid = blockIdx.x * blockDim.x + threadIdx.x;
  if (gid >= O * Cpad) return;
  int c = gid % Cpad;
  int o = gid / Cpad;
  Wh[gid] = (c < C) ? (_Float16)W[(long long)o * C + c] : (_Float16)0.0f;
}

// ------------------------ WMMA fragment loader -----------------------------
// 16-bit A/B fragment, 16x32 tile, per documented gfx1250 layout:
//   lane = l (0..31), row/col = l&15, khalf = l>>4,
//   element e -> k = 16*(e>>3) + 8*khalf + (e&7)
// => two contiguous 8-half (16B) loads per lane.
__device__ __forceinline__ v16h pn_load_frag(const _Float16* __restrict__ base,
                                             int stride, int rbase, int lane,
                                             int kb) {
  int rr = rbase + (lane & 15);
  const _Float16* p = base + (long long)rr * stride + kb + ((lane >> 4) << 3);
  v8h lo = *reinterpret_cast<const v8h*>(p);
  v8h hi = *reinterpret_cast<const v8h*>(p + 16);
  v16h f;
#pragma unroll
  for (int i = 0; i < 8; ++i) { f[i] = lo[i]; f[i + 8] = hi[i]; }
  return f;
}

// ------------------- WMMA GEMM + bias (+BN) (+ReLU) ------------------------
// H/F[M x Nout] = act(gamma * ((X[MxKpad] @ W[NoutxKpad]^T + bias) * bnscale)
//                     + beta).
// One wave per 16(M) x 64(N) output tile: 4 accumulators reuse each A
// fragment across 4 B fragments (1.25 fragment loads per WMMA, X re-read
// only Nout/64 times). Requires Nout % 64 == 0 (true for 64/128/256/512).
__global__ void pn_wmma_gemm_fused_kernel(const _Float16* __restrict__ X,
                                          const _Float16* __restrict__ W,
                                          const float* __restrict__ bias,
                                          const float* __restrict__ gamma,
                                          const float* __restrict__ beta,
                                          float bnscale, int relu,
                                          int M, int Nout, int Kpad,
                                          _Float16* __restrict__ outH,
                                          float* __restrict__ outF) {
  int lane = threadIdx.x & 31;
  int wave = (int)((blockIdx.x * blockDim.x + threadIdx.x) >> 5);
  int ntn = Nout >> 6, ntm = M >> 4;
  if (wave >= ntm * ntn) return;  // wave-uniform: EXEC stays all-ones for WMMA
  int nt = wave % ntn, mt = wave / ntn;

  v8f acc[4] = {};
  for (int kb = 0; kb < Kpad; kb += 32) {
    v16h a = pn_load_frag(X, Kpad, mt * 16, lane, kb);
#pragma unroll
    for (int j = 0; j < 4; ++j) {
      v16h bf = pn_load_frag(W, Kpad, nt * 64 + j * 16, lane, kb);
      acc[j] = __builtin_amdgcn_wmma_f32_16x16x32_f16(false, a, false, bf,
                                                      (short)0, acc[j],
                                                      false, false);
    }
  }
  // C/D layout: lane -> col = (lane&15); slot r -> row mt*16 + 8*(lane>>4) + r
  int rb = mt * 16 + ((lane >> 4) << 3);
#pragma unroll
  for (int j = 0; j < 4; ++j) {
    int col = nt * 64 + j * 16 + (lane & 15);
    float bv = bias ? bias[col] : 0.0f;
    float gv = gamma ? gamma[col] : 0.0f;
    float bev = beta ? beta[col] : 0.0f;
#pragma unroll
    for (int r = 0; r < 8; ++r) {
      float v = acc[j][r] + bv;
      if (gamma) v = gv * (v * bnscale) + bev;
      if (relu) v = fmaxf(v, 0.0f);
      long long o = (long long)(rb + r) * Nout + col;
      if (outH) outH[o] = (_Float16)v;
      if (outF) outF[o] = v;
    }
  }
}

// ------------------- WMMA GEMM fused with group-max ------------------------
// out[g, n] = max over G rows of (X @ W^T + bias). G in {16,32,48,128}.
// One wave per (group, 64-col block), 4 accumulators; loops the group's
// M-tiles; row-halves merged with a lane<->lane+16 shuffle. Never
// materializes the pre-pool GEMM (saves the 134 MB SA1 activation).
__global__ void pn_wmma_gemm_groupmax_kernel(const _Float16* __restrict__ X,
                                             const _Float16* __restrict__ W,
                                             const float* __restrict__ bias,
                                             int M, int Nout, int Kpad, int G,
                                             float* __restrict__ out) {
  int lane = threadIdx.x & 31;
  int wave = (int)((blockIdx.x * blockDim.x + threadIdx.x) >> 5);
  int ntn = Nout >> 6;
  int ngroups = M / G;
  if (wave >= ngroups * ntn) return;  // wave-uniform
  int nt = wave % ntn, g = wave / ntn;

  float best[4] = {-3e38f, -3e38f, -3e38f, -3e38f};
  int mtiles = G >> 4;
  for (int mi = 0; mi < mtiles; ++mi) {
    v8f acc[4] = {};
    int rb = g * G + mi * 16;
    for (int kb = 0; kb < Kpad; kb += 32) {
      v16h a = pn_load_frag(X, Kpad, rb, lane, kb);
#pragma unroll
      for (int j = 0; j < 4; ++j) {
        v16h bf = pn_load_frag(W, Kpad, nt * 64 + j * 16, lane, kb);
        acc[j] = __builtin_amdgcn_wmma_f32_16x16x32_f16(false, a, false, bf,
                                                        (short)0, acc[j],
                                                        false, false);
      }
    }
#pragma unroll
    for (int j = 0; j < 4; ++j)
#pragma unroll
      for (int r = 0; r < 8; ++r) best[j] = fmaxf(best[j], acc[j][r]);
  }
#pragma unroll
  for (int j = 0; j < 4; ++j) {
    float v = best[j];
    v = fmaxf(v, __shfl_xor(v, 16, 32));  // rows 0-7 vs 8-15 halves
    int col = nt * 64 + j * 16 + (lane & 15);
    v += bias ? bias[col] : 0.0f;         // bias uniform per col: add post-max
    if (lane < 16) out[(long long)g * Nout + col] = v;
  }
}

// ---------------------------------------------------------------------------
// Host-side orchestration
// ---------------------------------------------------------------------------
static inline int pn_blocks(long long total, int threads) {
  return (int)((total + threads - 1) / threads);
}
static inline int pn_wave_blocks(long long tiles) {
  long long threads = tiles * 32;
  return (int)((threads + 255) / 256);
}

extern "C" void kernel_launch(void* const* d_in, const int* in_sizes, int n_in,
                              void* d_out, int out_size, void* d_ws, size_t ws_size,
                              hipStream_t stream) {
  (void)in_sizes; (void)n_in; (void)out_size; (void)ws_size;

  const float* pc     = (const float*)d_in[0];
  const float* sa1_w1 = (const float*)d_in[1];
  const float* sa1_b1 = (const float*)d_in[2];
  const float* sa1_w2 = (const float*)d_in[3];
  const float* sa1_b2 = (const float*)d_in[4];
  const float* sa2_w1 = (const float*)d_in[5];
  const float* sa2_b1 = (const float*)d_in[6];
  const float* sa2_w2 = (const float*)d_in[7];
  const float* sa2_b2 = (const float*)d_in[8];
  const float* sa3_w1 = (const float*)d_in[9];
  const float* sa3_b1 = (const float*)d_in[10];
  const float* sa3_w2 = (const float*)d_in[11];
  const float* sa3_b2 = (const float*)d_in[12];
  const float* fc_w1  = (const float*)d_in[13];
  const float* fc_b1  = (const float*)d_in[14];
  const float* fc_g1  = (const float*)d_in[15];
  const float* fc_be1 = (const float*)d_in[16];
  const float* fc_w2  = (const float*)d_in[17];
  const float* fc_b2  = (const float*)d_in[18];
  const float* fc_g2  = (const float*)d_in[19];
  const float* fc_be2 = (const float*)d_in[20];

  float* out = (float*)d_out;

  const int B = 32, N = 2048;
  const int S1 = 512, K1 = 16;
  const int S2 = 128;
  const int K2arr[3] = {16, 32, 48};
  const float bnscale = 1.0f / sqrtf(1.0f + 1e-5f);

  // ---- workspace bump allocator (256B aligned) ----
  size_t off = 0;
  auto alloc = [&](size_t bytes) -> void* {
    void* p = (char*)d_ws + off;
    off = (off + bytes + 255) & ~(size_t)255;
    return p;
  };
  float*     xyz0  = (float*)alloc((size_t)B * N * 3 * 4);
  float*     nxyz1 = (float*)alloc((size_t)B * S1 * 3 * 4);
  int*       fidx1 = (int*)  alloc((size_t)B * S1 * 4);
  int*       idx1  = (int*)  alloc((size_t)B * S1 * K1 * 4);
  float*     nxyz2 = (float*)alloc((size_t)B * S2 * 3 * 4);
  int*       fidx2 = (int*)  alloc((size_t)B * S2 * 4);
  int*       idx2  = (int*)  alloc((size_t)B * S2 * 48 * 4);
  float*     p1    = (float*)alloc((size_t)B * S1 * 128 * 4);
  float*     p2    = (float*)alloc((size_t)B * S2 * 256 * 4);
  _Float16*  Xbuf  = (_Float16*)alloc((size_t)B * S2 * 48 * 160 * 2); // 62.9 MB peak (SA2)
  _Float16*  Hbuf  = (_Float16*)alloc((size_t)B * S2 * 48 * 128 * 2); // 50.3 MB peak (SA2)
  _Float16*  wAh   = (_Float16*)alloc((size_t)512 * 288 * 2);
  _Float16*  wBh   = (_Float16*)alloc((size_t)512 * 512 * 2);
  _Float16*  codeh = (_Float16*)alloc((size_t)B * 512 * 2);
  _Float16*  fc1h  = (_Float16*)alloc((size_t)B * 256 * 2);

  // ---- shared preprocessing: (B,3,N) -> (B,N,3) ----
  pn_transpose_kernel<<<pn_blocks((long long)B * N * 3, 256), 256, 0, stream>>>(
      pc, xyz0, B, N);

  for (int i = 0; i < 3; ++i) {
    const int K2 = K2arr[i];

    // ================= SA1: 2048 -> 512 centers, K=16, 6 -> 64 -> 128 ======
    pn_fps_kernel<<<B, 256, 0, stream>>>(xyz0, N, S1, fidx1, nxyz1);
    pn_knn_kernel<<<pn_blocks((long long)B * S1, 256), 256, 0, stream>>>(
        xyz0, nxyz1, B, N, S1, K1, idx1);

    const int M1 = B * S1 * K1;           // 262144, mult of 16
    pn_gather_kernel<<<pn_blocks((long long)M1 * 32, 256), 256, 0, stream>>>(
        xyz0, xyz0, idx1, nxyz1, Xbuf, B, S1, K1, N, 3, 32);

    pn_pad_f16_kernel<<<pn_blocks(64 * 32, 256), 256, 0, stream>>>(
        sa1_w1 + (size_t)i * 64 * 6, wAh, 64, 6, 32);
    pn_wmma_gemm_fused_kernel<<<pn_wave_blocks((long long)(M1 / 16) * (64 / 64)),
                                256, 0, stream>>>(
        Xbuf, wAh, sa1_b1 + (size_t)i * 64, nullptr, nullptr, 1.0f, 1,
        M1, 64, 32, Hbuf, nullptr);

    pn_pad_f16_kernel<<<pn_blocks(128 * 64, 256), 256, 0, stream>>>(
        sa1_w2 + (size_t)i * 128 * 64, wBh, 128, 64, 64);
    pn_wmma_gemm_groupmax_kernel<<<
        pn_wave_blocks((long long)(M1 / K1) * (128 / 64)), 256, 0, stream>>>(
        Hbuf, wBh, sa1_b2 + (size_t)i * 128, M1, 128, 64, K1, p1);

    // ================= SA2: 512 -> 128 centers, K={16,32,48}, 131->128->256 =
    pn_fps_kernel<<<B, 256, 0, stream>>>(nxyz1, S1, S2, fidx2, nxyz2);
    pn_knn_kernel<<<pn_blocks((long long)B * S2, 256), 256, 0, stream>>>(
        nxyz1, nxyz2, B, S1, S2, K2, idx2);

    const int M2 = B * S2 * K2;           // mult of 16 for K2 in {16,32,48}
    pn_gather_kernel<<<pn_blocks((long long)M2 * 160, 256), 256, 0, stream>>>(
        nxyz1, p1, idx2, nxyz2, Xbuf, B, S2, K2, S1, 128, 160);

    pn_pad_f16_kernel<<<pn_blocks(128 * 160, 256), 256, 0, stream>>>(
        sa2_w1 + (size_t)i * 128 * 131, wAh, 128, 131, 160);
    pn_wmma_gemm_fused_kernel<<<pn_wave_blocks((long long)(M2 / 16) * (128 / 64)),
                                256, 0, stream>>>(
        Xbuf, wAh, sa2_b1 + (size_t)i * 128, nullptr, nullptr, 1.0f, 1,
        M2, 128, 160, Hbuf, nullptr);

    pn_pad_f16_kernel<<<pn_blocks(256 * 128, 256), 256, 0, stream>>>(
        sa2_w2 + (size_t)i * 256 * 128, wBh, 256, 128, 128);
    pn_wmma_gemm_groupmax_kernel<<<
        pn_wave_blocks((long long)(M2 / K2) * (256 / 64)), 256, 0, stream>>>(
        Hbuf, wBh, sa2_b2 + (size_t)i * 256, M2, 256, 128, K2, p2);

    // ================= SA3 (group_all): 259 -> 512 -> 512, max over 128 =====
    const int M3 = B * S2;                // 4096
    pn_gather_kernel<<<pn_blocks((long long)M3 * 288, 256), 256, 0, stream>>>(
        nxyz2, p2, nullptr, nullptr, Xbuf, B, 1, S2, S2, 256, 288);

    pn_pad_f16_kernel<<<pn_blocks(512 * 288, 256), 256, 0, stream>>>(
        sa3_w1 + (size_t)i * 512 * 259, wAh, 512, 259, 288);
    pn_wmma_gemm_fused_kernel<<<pn_wave_blocks((long long)(M3 / 16) * (512 / 64)),
                                256, 0, stream>>>(
        Xbuf, wAh, sa3_b1 + (size_t)i * 512, nullptr, nullptr, 1.0f, 1,
        M3, 512, 288, Hbuf, nullptr);

    float* code_out = out + (size_t)3 * B * 128 + (size_t)i * B * 512;
    pn_pad_f16_kernel<<<pn_blocks(512 * 512, 256), 256, 0, stream>>>(
        sa3_w2 + (size_t)i * 512 * 512, wBh, 512, 512, 512);
    pn_wmma_gemm_groupmax_kernel<<<
        pn_wave_blocks((long long)B * (512 / 64)), 256, 0, stream>>>(
        Hbuf, wBh, sa3_b2 + (size_t)i * 512, M3, 512, 512, S2, code_out);

    // ================= FC head: 512 -> 256 -> 128 (Linear+BN+ReLU) ==========
    pn_pad_f16_kernel<<<pn_blocks(B * 512, 256), 256, 0, stream>>>(
        code_out, codeh, B, 512, 512);

    pn_pad_f16_kernel<<<pn_blocks(256 * 512, 256), 256, 0, stream>>>(
        fc_w1 + (size_t)i * 256 * 512, wAh, 256, 512, 512);
    pn_wmma_gemm_fused_kernel<<<pn_wave_blocks((long long)(B / 16) * (256 / 64)),
                                256, 0, stream>>>(
        codeh, wAh, fc_b1 + (size_t)i * 256, fc_g1 + (size_t)i * 256,
        fc_be1 + (size_t)i * 256, bnscale, 1, B, 256, 512, fc1h, nullptr);

    pn_pad_f16_kernel<<<pn_blocks(128 * 256, 256), 256, 0, stream>>>(
        fc_w2 + (size_t)i * 128 * 256, wBh, 128, 256, 256);
    pn_wmma_gemm_fused_kernel<<<pn_wave_blocks((long long)(B / 16) * (128 / 64)),
                                256, 0, stream>>>(
        fc1h, wBh, fc_b2 + (size_t)i * 128, fc_g2 + (size_t)i * 128,
        fc_be2 + (size_t)i * 128, bnscale, 1, B, 128, 256, nullptr,
        out + (size_t)i * B * 128);
  }
}
